// ContrastiveLoss_10247791968969
// MI455X (gfx1250) — matrix operand order repaired and checked by reading, not compiled, not run on hardware
//
#include <hip/hip_runtime.h>
#include <hip/hip_bf16.h>
#include <math.h>

#define BATCH 4096
#define TWOB  8192
#define DIM   512
#define INV_TEMP 2.0f          // 1 / 0.5
#define NTILES (TWOB / 16)     // 512 column tiles
#define MTILES (TWOB / 16)     // 512 row tiles

typedef __attribute__((ext_vector_type(16))) __bf16 v16bf;
typedef __attribute__((ext_vector_type(8)))  float  v8f;

union FragU { uint4 u[2]; v16bf v; };

// ---------------------------------------------------------------------------
// Kernel 1: row-wise L2 normalization, fp32 -> bf16, writing reps[8192][512].
// One wave (32 lanes) per row; each lane handles 16 contiguous floats.
// ---------------------------------------------------------------------------
__global__ void l2norm_to_bf16_kernel(const float* __restrict__ emb_i,
                                      const float* __restrict__ emb_j,
                                      __hip_bfloat16* __restrict__ reps)
{
    const int row  = (blockIdx.x * blockDim.x + threadIdx.x) >> 5;
    const int lane = threadIdx.x & 31;
    if (row >= TWOB) return;

    const float* src = (row < BATCH) ? emb_i + (size_t)row * DIM
                                     : emb_j + (size_t)(row - BATCH) * DIM;
    float4 v[4];
    float ss = 0.0f;
#pragma unroll
    for (int i = 0; i < 4; ++i) {
        v[i] = ((const float4*)src)[lane * 4 + i];
        ss += v[i].x * v[i].x + v[i].y * v[i].y + v[i].z * v[i].z + v[i].w * v[i].w;
    }
#pragma unroll
    for (int m = 16; m >= 1; m >>= 1) ss += __shfl_xor(ss, m, 32);

    const float scale = 1.0f / fmaxf(sqrtf(ss), 1e-12f);
    __hip_bfloat16* dst = reps + (size_t)row * DIM + lane * 16;
#pragma unroll
    for (int i = 0; i < 4; ++i) {
        dst[i * 4 + 0] = __float2bfloat16(v[i].x * scale);
        dst[i * 4 + 1] = __float2bfloat16(v[i].y * scale);
        dst[i * 4 + 2] = __float2bfloat16(v[i].z * scale);
        dst[i * 4 + 3] = __float2bfloat16(v[i].w * scale);
    }
}

// ---------------------------------------------------------------------------
// Kernel 2: fused sim + exp + row-sum + positive capture.
// One wave owns a 16-row tile (A kept resident in 128 VGPRs across full K=512)
// and sweeps all 512 column tiles with v_wmma_f32_16x16x32_bf16.
// B fragments flow through a 4-deep software-pipelined ring (3 load-pairs in
// flight at each consumption point) that also pipelines across tile borders.
// ---------------------------------------------------------------------------
__global__ void __launch_bounds__(256)
nt_xent_rows_kernel(const __hip_bfloat16* __restrict__ reps,
                    float* __restrict__ partials)
{
    const int lane      = threadIdx.x & 31;
    const int waveInBlk = threadIdx.x >> 5;              // 0..7
    const int tile      = blockIdx.x * 8 + waveInBlk;    // 0..511
    const int m0        = tile * 16;
    const int lm        = lane & 15;                     // N index / row-in-tile
    const int khalf     = lane >> 4;                     // K-half select

    // ---- Load resident A fragments: row m0+lm, per-lane contiguous K layout (k = 16h + s) ----
    v16bf a[16];
    const __hip_bfloat16* arow = reps + (size_t)(m0 + lm) * DIM;
#pragma unroll
    for (int kt = 0; kt < 16; ++kt) {
        FragU f;
        f.u[0] = *(const uint4*)(arow + kt * 32 + khalf * 16);      // K [16h, 16h+8)
        f.u[1] = *(const uint4*)(arow + kt * 32 + khalf * 16 + 8);  // K [16h+8, +8)
        a[kt] = f.v;
    }

    float rowsum[8], pos[8];
#pragma unroll
    for (int r = 0; r < 8; ++r) { rowsum[r] = 0.0f; pos[r] = 0.0f; }

    // Per-lane B base: column row (j0+lm), starting at K element khalf*16.
    const __hip_bfloat16* bbase = reps + (size_t)lm * DIM + khalf * 16;

    // ---- Prologue: issue B fragments 0..2 of tile 0 into the ring ----
    FragU b[4];
#pragma unroll
    for (int p = 0; p < 3; ++p) {
        b[p].u[0] = *(const uint4*)(bbase + p * 32);
        b[p].u[1] = *(const uint4*)(bbase + p * 32 + 8);
    }

    for (int jt = 0; jt < NTILES; ++jt) {
        const int j0 = jt * 16;
        const __hip_bfloat16* brow_cur = bbase + (size_t)j0 * DIM;
        const __hip_bfloat16* brow_nxt =
            bbase + (size_t)(((jt + 1) & (NTILES - 1)) * 16) * DIM;

        // Prefetch the next tile's B rows into cache.
        __builtin_prefetch(brow_nxt, 0, 0);

        v8f c = {0.f, 0.f, 0.f, 0.f, 0.f, 0.f, 0.f, 0.f};
#pragma unroll
        for (int kt = 0; kt < 16; ++kt) {
            // Issue the load for fragment kt+3 (spills into the next tile).
            const int pf = kt + 3;
            const __hip_bfloat16* p = (pf < 16) ? (brow_cur + pf * 32)
                                                : (brow_nxt + (pf - 16) * 32);
            b[pf & 3].u[0] = *(const uint4*)p;
            b[pf & 3].u[1] = *(const uint4*)(p + 8);

            // Consume fragment kt (loaded 3 iterations ago).
            c = __builtin_amdgcn_wmma_f32_16x16x32_bf16(
                    false, a[kt], false, b[kt & 3].v, (short)0, c, false, false);
        }

        // Epilogue: c[r] = sim[m0 + r + 8*khalf][j0 + lm]
#pragma unroll
        for (int r = 0; r < 8; ++r) {
            const float s    = c[r];
            const int   grow = m0 + r + khalf * 8;
            const int   gcol = j0 + lm;
            float e = __expf(s * INV_TEMP);        // exp(sim / temp), |s|<=~1
            if (gcol == grow) e = 0.0f;            // exclude diagonal
            rowsum[r] += e;
            const int pcol = (grow < BATCH) ? grow + BATCH : grow - BATCH;
            if (gcol == pcol) pos[r] = s;          // capture positive
        }
    }

    // ---- Reduce row sums / positives across the 16 lanes of each half ----
    float myloss = 0.0f;
#pragma unroll
    for (int r = 0; r < 8; ++r) {
        float rs = rowsum[r];
        float p  = pos[r];
#pragma unroll
        for (int m = 8; m >= 1; m >>= 1) {
            rs += __shfl_xor(rs, m, 32);
            p  += __shfl_xor(p,  m, 32);
        }
        if (lm == 0) myloss += logf(rs) - p * INV_TEMP;  // log(denom) - pos/temp
    }
    myloss += __shfl_xor(myloss, 16, 32);  // lane0 gets both halves (16 rows)

    __shared__ float blksum[8];
    if (lane == 0) blksum[waveInBlk] = myloss;
    __syncthreads();
    if (threadIdx.x == 0) {
        float s = 0.0f;
#pragma unroll
        for (int w = 0; w < 8; ++w) s += blksum[w];
        partials[blockIdx.x] = s;
    }
}

// ---------------------------------------------------------------------------
// Kernel 3: deterministic final reduction of the 64 block partials.
// ---------------------------------------------------------------------------
__global__ void final_reduce_kernel(const float* __restrict__ partials,
                                    int n, float* __restrict__ out)
{
    if (blockIdx.x == 0 && threadIdx.x == 0) {
        float s = 0.0f;
        for (int i = 0; i < n; ++i) s += partials[i];
        out[0] = s / (float)TWOB;   // mean over 2B rows
    }
}

// ---------------------------------------------------------------------------
extern "C" void kernel_launch(void* const* d_in, const int* in_sizes, int n_in,
                              void* d_out, int out_size, void* d_ws, size_t ws_size,
                              hipStream_t stream)
{
    (void)in_sizes; (void)n_in; (void)out_size; (void)ws_size;

    const float* emb_i = (const float*)d_in[0];
    const float* emb_j = (const float*)d_in[1];
    float* out = (float*)d_out;

    // Workspace layout: reps bf16 [8192][512] (8 MB), then 64 float partials.
    __hip_bfloat16* reps = (__hip_bfloat16*)d_ws;
    float* partials = (float*)((char*)d_ws + (size_t)TWOB * DIM * sizeof(__hip_bfloat16));

    // 1) normalize: one wave per row -> 8192 waves = 1024 blocks of 256
    l2norm_to_bf16_kernel<<<TWOB / 8, 256, 0, stream>>>(emb_i, emb_j, reps);

    // 2) fused sim/softmax-denominator sweep: 512 row tiles, 8 waves/block
    const int nblocks = MTILES / 8;  // 64
    nt_xent_rows_kernel<<<nblocks, 256, 0, stream>>>(reps, partials);

    // 3) deterministic scalar reduction
    final_reduce_kernel<<<1, 32, 0, stream>>>(partials, nblocks, out);
}
